// target_knn_84645215470249
// MI455X (gfx1250) — matrix operand order repaired and checked by reading, not compiled, not run on hardware
//
#include <hip/hip_runtime.h>
#include <hip/hip_bf16.h>

// ---------------------------------------------------------------------------
// Problem constants (match reference)
// ---------------------------------------------------------------------------
#define NLP   16384      // lossy points
#define NRP   16384      // ref points
#define KNN_K 16         // neighbors
#define CCH   256        // channels
#define EPS_BN 1e-5f

#define KNN_CHUNKS 8
#define KNN_CHSZ   (NRP / KNN_CHUNKS)   // 2048 refs per chunk

typedef __attribute__((ext_vector_type(16))) _Float16 v16h;
typedef __attribute__((ext_vector_type(8)))  _Float16 v8h;
typedef __attribute__((ext_vector_type(8)))  float    v8f;

struct alignas(16) NbrHdr { int g; float r0, r1, r2; };

// ---------------------------------------------------------------------------
// KNN stage 1: each (point, ref-chunk) pair keeps a sorted partial top-16.
// Grid (NL/256, KNN_CHUNKS): 4096 waves -> fills the WGPs.
// Partials stored transposed [slot][point] for coalesced merge reads.
// ---------------------------------------------------------------------------
__global__ __launch_bounds__(256) void knn_partial_kernel(const float* __restrict__ lc,
                                                          const float* __restrict__ rc,
                                                          float* __restrict__ pd,
                                                          int*   __restrict__ pi) {
    __shared__ float sx[256], sy[256], sz[256];
    const int n  = blockIdx.x * 256 + threadIdx.x;
    const int ch = blockIdx.y;
    const float px = lc[n * 3 + 0];
    const float py = lc[n * 3 + 1];
    const float pz = lc[n * 3 + 2];

    float bd[KNN_K];
    int   bi[KNN_K];
#pragma unroll
    for (int i = 0; i < KNN_K; ++i) { bd[i] = 3.4e38f; bi[i] = 0; }

    const int t0 = ch * KNN_CHSZ;
    for (int t = t0; t < t0 + KNN_CHSZ; t += 256) {
        const int j = t + threadIdx.x;
        sx[threadIdx.x] = rc[j * 3 + 0];
        sy[threadIdx.x] = rc[j * 3 + 1];
        sz[threadIdx.x] = rc[j * 3 + 2];
        if (t + 256 < t0 + KNN_CHSZ) {   // global_prefetch_b8 on next tile
            __builtin_prefetch(&rc[(t + 256) * 3 + threadIdx.x], 0, 1);
        }
        __syncthreads();
#pragma unroll 4
        for (int j2 = 0; j2 < 256; ++j2) {
            const float dx = px - sx[j2];
            const float dy = py - sy[j2];
            const float dz = pz - sz[j2];
            const float d2 = dx * dx + dy * dy + dz * dz;
            if (d2 < bd[KNN_K - 1]) {
                const int gi = t + j2;
#pragma unroll
                for (int q = KNN_K - 1; q >= 1; --q) {
                    if (d2 < bd[q - 1])  { bd[q] = bd[q - 1]; bi[q] = bi[q - 1]; }
                    else if (d2 < bd[q]) { bd[q] = d2;        bi[q] = gi; }
                }
                if (d2 < bd[0])          { bd[0] = d2;        bi[0] = gi; }
            }
        }
        __syncthreads();
    }
#pragma unroll
    for (int q = 0; q < KNN_K; ++q) {
        const long slot = (long)(ch * KNN_K + q) * NLP + n;  // transposed
        pd[slot] = bd[q];
        pi[slot] = bi[q];
    }
}

// ---------------------------------------------------------------------------
// KNN stage 2: merge 8 sorted partial lists (128 candidates) -> final top-16,
// emit neighbor headers {ref index, relative xyz}.
// ---------------------------------------------------------------------------
__global__ __launch_bounds__(256) void knn_merge_kernel(const float* __restrict__ pd,
                                                        const int*   __restrict__ pi,
                                                        const float* __restrict__ lc,
                                                        const float* __restrict__ rc,
                                                        NbrHdr* __restrict__ hdr) {
    const int n = blockIdx.x * 256 + threadIdx.x;
    float bd[KNN_K];
    int   bi[KNN_K];
#pragma unroll
    for (int i = 0; i < KNN_K; ++i) { bd[i] = 3.4e38f; bi[i] = 0; }

    for (int t = 0; t < KNN_CHUNKS * KNN_K; ++t) {
        const float d2 = pd[(long)t * NLP + n];
        const int   gi = pi[(long)t * NLP + n];
        if (d2 < bd[KNN_K - 1]) {
#pragma unroll
            for (int q = KNN_K - 1; q >= 1; --q) {
                if (d2 < bd[q - 1])  { bd[q] = bd[q - 1]; bi[q] = bi[q - 1]; }
                else if (d2 < bd[q]) { bd[q] = d2;        bi[q] = gi; }
            }
            if (d2 < bd[0])          { bd[0] = d2;        bi[0] = gi; }
        }
    }

    const float px = lc[n * 3 + 0];
    const float py = lc[n * 3 + 1];
    const float pz = lc[n * 3 + 2];
#pragma unroll
    for (int k = 0; k < KNN_K; ++k) {
        const int g = bi[k];
        NbrHdr h;
        h.g  = g;
        h.r0 = px - rc[g * 3 + 0];
        h.r1 = py - rc[g * 3 + 1];
        h.r2 = pz - rc[g * 3 + 2];
        hdr[n * KNN_K + k] = h;
    }
}

// ---------------------------------------------------------------------------
// WMMA GEMM: Y[M,256] = X[M,256] @ W[256,256] + bias  (f16 math, f32 accum)
// 512 threads = 16 waves; WG tile 128 rows x 256 cols; K-step 32.
// Wave w: row stripe (w&7)*16, column half (w>>3)*128 -> 8 acc tiles/wave
// (64 VGPRs) so the allocator can double-buffer B fragments and occupancy
// reaches ~6 waves/SIMD to hide LDS/WMMA latency.
//  - Xs row-major  [128][40]  : A fragment = 2 aligned ds_load_b128
//  - Wt col-major  [256][40]  : B fragment = 2 aligned ds_load_b128
//  - next k-step's global reads register-staged to overlap HBM with WMMA
// ---------------------------------------------------------------------------
#define XS_STR 40     // halves; 80B rows -> 16B-aligned b128 fragment loads
#define WT_STR 40
#define GEMM_THREADS 512

struct PlainX {
    const float* X;
    __device__ float operator()(long m, int k) const { return X[m * CCH + k]; }
};
struct GatherPosX {  // ref_F[g] + rel @ pos_w + pos_b   (fused gather + pos-MLP)
    const NbrHdr* hdr; const float* refF; const float* pw; const float* pb;
    __device__ float operator()(long m, int k) const {
        const NbrHdr h = hdr[m];
        float v = refF[(long)h.g * CCH + k];
        v += h.r0 * pw[0 * CCH + k];
        v += h.r1 * pw[1 * CCH + k];
        v += h.r2 * pw[2 * CCH + k];
        return v + pb[k];
    }
};
struct StoreF32 {
    float* Y;
    __device__ void operator()(long m, int c, float v) const { Y[m * CCH + c] = v; }
};
struct StoreF16 {
    _Float16* Y;
    __device__ void operator()(long m, int c, float v) const { Y[m * CCH + c] = (_Float16)v; }
};

template <typename XF, typename YS>
__global__ __launch_bounds__(GEMM_THREADS) void gemm256_kernel(XF xf,
                                                               const float* __restrict__ W,
                                                               const float* __restrict__ bias,
                                                               YS ys) {
    __shared__ _Float16 Xs[128 * XS_STR];
    __shared__ _Float16 Wt[256 * WT_STR];

    const int  tid  = threadIdx.x;
    const int  wave = tid >> 5;          // wave32: 16 waves
    const int  lane = tid & 31;
    const int  half = lane >> 4;
    const int  l16  = lane & 15;
    const int  wr8  = wave & 7;          // row stripe within the 128-row tile
    const int  colg = (wave >> 3) * 128; // column half owned by this wave
    const long row0 = (long)blockIdx.x * 128;

    const v8f vzero = {0.f, 0.f, 0.f, 0.f, 0.f, 0.f, 0.f, 0.f};
    v8f acc[8];
#pragma unroll
    for (int t = 0; t < 8; ++t) acc[t] = vzero;

    float xr[8];    // staged X elements (this thread's share of the tile)
    float wr[16];   // staged W elements

    auto fetch = [&](int kk) {
#pragma unroll
        for (int i = 0; i < 8; ++i) {
            const int e = tid + i * GEMM_THREADS;         // 128*32 elements
            xr[i] = xf(row0 + (e >> 5), kk + (e & 31));
        }
#pragma unroll
        for (int i = 0; i < 16; ++i) {
            const int e = tid + i * GEMM_THREADS;         // 32*256, n fast
            wr[i] = W[(kk + (e >> 8)) * CCH + (e & 255)];
        }
    };

    auto loadB = [&](int nt) -> v16h {
        const _Float16* wcol = &Wt[(colg + nt * 16 + l16) * WT_STR];
        const v8h b0 = *(const v8h*)&wcol[8 * half];
        const v8h b1 = *(const v8h*)&wcol[16 + 8 * half];
        return __builtin_shufflevector(b0, b1, 0, 1, 2, 3, 4, 5, 6, 7,
                                       8, 9, 10, 11, 12, 13, 14, 15);
    };

    fetch(0);

#pragma unroll 1
    for (int step = 0; step < 8; ++step) {
        if (step) __syncthreads();       // previous tile fully consumed
        // commit staged registers to LDS (f32 -> f16)
#pragma unroll
        for (int i = 0; i < 8; ++i) {
            const int e = tid + i * GEMM_THREADS;
            Xs[(e >> 5) * XS_STR + (e & 31)] = (_Float16)xr[i];
        }
#pragma unroll
        for (int i = 0; i < 16; ++i) {
            const int e = tid + i * GEMM_THREADS;
            Wt[(e & 255) * WT_STR + (e >> 8)] = (_Float16)wr[i];   // transposed
        }
        __syncthreads();

        if (step < 7) fetch((step + 1) * 32);   // overlap next HBM reads w/ WMMA

        // ---- A fragment: lane row = wr8*16+l16; two contiguous 8-half runs
        const _Float16* xrow = &Xs[(wr8 * 16 + l16) * XS_STR];
        const v8h a0 = *(const v8h*)&xrow[8 * half];        // k = 8h .. 8h+7
        const v8h a1 = *(const v8h*)&xrow[16 + 8 * half];   // k = 16+8h .. 16+8h+7
        const v16h a = __builtin_shufflevector(a0, a1, 0, 1, 2, 3, 4, 5, 6, 7,
                                               8, 9, 10, 11, 12, 13, 14, 15);

        // ---- 8 column tiles, B fragment pipelined one tile ahead ----
        v16h bcur = loadB(0);
#pragma unroll
        for (int nt = 0; nt < 8; ++nt) {
            const v16h bnxt = (nt < 7) ? loadB(nt + 1) : bcur;
            acc[nt] = __builtin_amdgcn_wmma_f32_16x16x32_f16(
                false, a, false, bcur, (short)0, acc[nt], false, false);
            bcur = bnxt;
        }
    }

    // ---- epilogue: bias + store (C layout: VGPR r -> row r + 8*half)
#pragma unroll
    for (int nt = 0; nt < 8; ++nt) {
        const int col = colg + nt * 16 + l16;
        const float bcol = bias[col];
#pragma unroll
        for (int r = 0; r < 8; ++r) {
            const long m = row0 + wr8 * 16 + half * 8 + r;
            ys(m, col, acc[nt][r] + bcol);
        }
    }
}

// ---------------------------------------------------------------------------
// Local attention: one wave per point; lane owns 8 *contiguous* channels so
// Km/V rows are single 16B f16x8 loads and the output stores merge to b128.
// ---------------------------------------------------------------------------
__global__ __launch_bounds__(256) void attn_kernel(const float* __restrict__ Q,
                                                   const _Float16* __restrict__ Km,
                                                   const _Float16* __restrict__ Vv,
                                                   float* __restrict__ out) {
    const int wave = threadIdx.x >> 5;
    const int lane = threadIdx.x & 31;
    const long n = (long)blockIdx.x * 8 + wave;
    const int cbase = lane * 8;

    float qv[8];
#pragma unroll
    for (int i = 0; i < 8; ++i) qv[i] = Q[n * CCH + cbase + i];

    float s[KNN_K];
#pragma unroll
    for (int k = 0; k < KNN_K; ++k) {
        const v8h kv = *(const v8h*)&Km[(n * KNN_K + k) * CCH + cbase];
        float p = 0.f;
#pragma unroll
        for (int i = 0; i < 8; ++i) p += qv[i] * (float)kv[i];
#pragma unroll
        for (int off = 16; off >= 1; off >>= 1) p += __shfl_xor(p, off, 32);
        s[k] = p * (1.0f / 16.0f);      // / sqrt(C)
    }

    float mx = s[0];
#pragma unroll
    for (int k = 1; k < KNN_K; ++k) mx = fmaxf(mx, s[k]);
    float e[KNN_K], sum = 0.f;
#pragma unroll
    for (int k = 0; k < KNN_K; ++k) { e[k] = __expf(s[k] - mx); sum += e[k]; }
    const float inv = 1.0f / sum;

    float o[8];
#pragma unroll
    for (int i = 0; i < 8; ++i) o[i] = 0.f;
#pragma unroll
    for (int k = 0; k < KNN_K; ++k) {
        const float w = e[k] * inv;
        const v8h vv = *(const v8h*)&Vv[(n * KNN_K + k) * CCH + cbase];
#pragma unroll
        for (int i = 0; i < 8; ++i) o[i] += w * (float)vv[i];
    }
#pragma unroll
    for (int i = 0; i < 8; ++i) out[n * CCH + cbase + i] = o[i];
}

// ---------------------------------------------------------------------------
// Training-mode BN over the point dim: deterministic two-stage reduction.
// ---------------------------------------------------------------------------
__global__ __launch_bounds__(256) void resid_stats_kernel(const float* __restrict__ A,
                                                          const float* __restrict__ B,
                                                          float* __restrict__ X,
                                                          float* __restrict__ part) {
    const int c = threadIdx.x;
    const int b = blockIdx.x;            // 256 blocks x 64 rows each
    float s = 0.f, sq = 0.f;
    for (int r = 0; r < 64; ++r) {
        const long i = ((long)b * 64 + r) * CCH + c;
        const float x = A[i] + B[i];
        X[i] = x;
        s += x; sq += x * x;
    }
    part[b * 512 + c]       = s;
    part[b * 512 + 256 + c] = sq;
}

__global__ __launch_bounds__(256) void finalize_stats_kernel(const float* __restrict__ part,
                                                             const float* __restrict__ g,
                                                             const float* __restrict__ bta,
                                                             float* __restrict__ ss) {
    const int c = threadIdx.x;
    float s = 0.f, sq = 0.f;
    for (int b = 0; b < 256; ++b) { s += part[b * 512 + c]; sq += part[b * 512 + 256 + c]; }
    const float mean = s  * (1.0f / (float)NLP);
    const float var  = sq * (1.0f / (float)NLP) - mean * mean;
    const float rs   = rsqrtf(var + EPS_BN);
    const float sc   = g[c] * rs;
    ss[c]       = sc;
    ss[256 + c] = bta[c] - mean * sc;
}

__global__ __launch_bounds__(256) void bn_apply_kernel(const float* __restrict__ X,
                                                       const float* __restrict__ ss,
                                                       float* __restrict__ Y) {
    const long i = (long)blockIdx.x * 256 + threadIdx.x;
    const int  c = (int)(i & (CCH - 1));
    Y[i] = X[i] * ss[c] + ss[256 + c];
}

// ---------------------------------------------------------------------------
// Host-side launch sequence
// ---------------------------------------------------------------------------
extern "C" void kernel_launch(void* const* d_in, const int* in_sizes, int n_in,
                              void* d_out, int out_size, void* d_ws, size_t ws_size,
                              hipStream_t stream) {
    const float* lossy_C = (const float*)d_in[0];
    const float* lossy_F = (const float*)d_in[1];
    const float* ref_C   = (const float*)d_in[2];
    const float* ref_F   = (const float*)d_in[3];
    const float* pos0_w  = (const float*)d_in[4];
    const float* pos0_b  = (const float*)d_in[5];
    const float* pos1_w  = (const float*)d_in[6];
    const float* pos1_b  = (const float*)d_in[7];
    const float* q0_w = (const float*)d_in[8];  const float* q0_b = (const float*)d_in[9];
    const float* k0_w = (const float*)d_in[10]; const float* k0_b = (const float*)d_in[11];
    const float* v0_w = (const float*)d_in[12]; const float* v0_b = (const float*)d_in[13];
    const float* q1_w = (const float*)d_in[14]; const float* q1_b = (const float*)d_in[15];
    const float* k1_w = (const float*)d_in[16]; const float* k1_b = (const float*)d_in[17];
    const float* v1_w = (const float*)d_in[18]; const float* v1_b = (const float*)d_in[19];
    const float* lin_w = (const float*)d_in[20]; const float* lin_b = (const float*)d_in[21];
    const float* bn0_g = (const float*)d_in[22]; const float* bn0_b = (const float*)d_in[23];
    const float* bn1_g = (const float*)d_in[24]; const float* bn1_b = (const float*)d_in[25];

    const long M = (long)NLP * KNN_K;   // 262144 gathered rows

    // ---- workspace carve-out (~340 MB peak) ----
    char* ws = (char*)d_ws;
    size_t off = 0;
    auto carve = [&](size_t bytes) -> char* {
        char* p = ws + off;
        off = (off + bytes + 255) & ~(size_t)255;
        return p;
    };
    NbrHdr*   hdr   = (NbrHdr*)  carve(M * sizeof(NbrHdr));                    //   4 MB
    float*    pd    = (float*)   carve((size_t)NLP * KNN_CHUNKS * KNN_K * 4);  //   8 MB
    int*      pi    = (int*)     carve((size_t)NLP * KNN_CHUNKS * KNN_K * 4);  //   8 MB
    float*    Qbuf  = (float*)   carve((size_t)NLP * CCH * 4);                 //  16 MB (also lin out)
    _Float16* Km    = (_Float16*)carve((size_t)M * CCH * 2);                   // 128 MB
    _Float16* Vv    = (_Float16*)carve((size_t)M * CCH * 2);                   // 128 MB
    float*    attnb = (float*)   carve((size_t)NLP * CCH * 4);                 //  16 MB (also z)
    float*    xbuf  = (float*)   carve((size_t)NLP * CCH * 4);                 //  16 MB
    float*    ybuf  = (float*)   carve((size_t)NLP * CCH * 4);                 //  16 MB
    float*    part  = (float*)   carve(256 * 512 * 4);
    float*    ssbuf = (float*)   carve(512 * 4);
    (void)ws_size; (void)in_sizes; (void)n_in; (void)out_size;

    const dim3 blk(256);
    const dim3 gblk(GEMM_THREADS);
    const int gemmBlocksSmall = NLP / 128;        // 128
    const int gemmBlocksBig   = (int)(M / 128);   // 2048

    // 1) KNN (8-way split + merge) -> neighbor headers (idx + relative xyz)
    knn_partial_kernel<<<dim3(NLP / 256, KNN_CHUNKS), blk, 0, stream>>>(lossy_C, ref_C, pd, pi);
    knn_merge_kernel<<<dim3(NLP / 256), blk, 0, stream>>>(pd, pi, lossy_C, ref_C, hdr);

    // ---- attention layer 0 ----
    gemm256_kernel<<<dim3(gemmBlocksSmall), gblk, 0, stream>>>(
        PlainX{lossy_F}, q0_w, q0_b, StoreF32{Qbuf});
    gemm256_kernel<<<dim3(gemmBlocksBig), gblk, 0, stream>>>(
        GatherPosX{hdr, ref_F, pos0_w, pos0_b}, k0_w, k0_b, StoreF16{Km});
    gemm256_kernel<<<dim3(gemmBlocksBig), gblk, 0, stream>>>(
        GatherPosX{hdr, ref_F, pos0_w, pos0_b}, v0_w, v0_b, StoreF16{Vv});
    attn_kernel<<<dim3(NLP / 8), blk, 0, stream>>>(Qbuf, Km, Vv, attnb);

    // ---- attention layer 1 ----
    gemm256_kernel<<<dim3(gemmBlocksSmall), gblk, 0, stream>>>(
        PlainX{attnb}, q1_w, q1_b, StoreF32{Qbuf});
    gemm256_kernel<<<dim3(gemmBlocksBig), gblk, 0, stream>>>(
        GatherPosX{hdr, ref_F, pos1_w, pos1_b}, k1_w, k1_b, StoreF16{Km});
    gemm256_kernel<<<dim3(gemmBlocksBig), gblk, 0, stream>>>(
        GatherPosX{hdr, ref_F, pos1_w, pos1_b}, v1_w, v1_b, StoreF16{Vv});
    attn_kernel<<<dim3(NLP / 8), blk, 0, stream>>>(Qbuf, Km, Vv, attnb);

    // ---- bn0(lossy_F + out) ----
    resid_stats_kernel<<<dim3(256), blk, 0, stream>>>(lossy_F, attnb, xbuf, part);
    finalize_stats_kernel<<<dim3(1), blk, 0, stream>>>(part, bn0_g, bn0_b, ssbuf);
    bn_apply_kernel<<<dim3(NLP), blk, 0, stream>>>(xbuf, ssbuf, ybuf);

    // ---- out1 = y @ lin_w + lin_b ; bn1(out1 + y) ----
    gemm256_kernel<<<dim3(gemmBlocksSmall), gblk, 0, stream>>>(
        PlainX{ybuf}, lin_w, lin_b, StoreF32{Qbuf});                 // reuse Qbuf
    resid_stats_kernel<<<dim3(256), blk, 0, stream>>>(Qbuf, ybuf, attnb, part); // attnb = z
    finalize_stats_kernel<<<dim3(1), blk, 0, stream>>>(part, bn1_g, bn1_b, ssbuf);
    bn_apply_kernel<<<dim3(NLP), blk, 0, stream>>>(attnb, ssbuf, (float*)d_out);
}
